// MutiheadAttention_43937515438531
// MI455X (gfx1250) — compile-verified, matrix operand order checked by reading, not run on hardware
//
#include <hip/hip_runtime.h>

typedef __attribute__((ext_vector_type(16))) __bf16 bf16x16;
typedef __attribute__((ext_vector_type(8)))  __bf16 bf16x8;
typedef __attribute__((ext_vector_type(8)))  float  f32x8;
typedef __attribute__((ext_vector_type(4)))  unsigned short u16x4;
typedef __attribute__((ext_vector_type(4)))  unsigned int   u32x4;
typedef __attribute__((ext_vector_type(8)))  unsigned int   u32x8;

#define B_   4
#define S_   2048
#define DIN  1024
#define H_   16
#define DH   64

// ---------- helpers ----------

__device__ __forceinline__ unsigned short f2bfu(float f) {
    unsigned u = __builtin_bit_cast(unsigned, f);
    return (unsigned short)((u + 0x7FFFu + ((u >> 16) & 1u)) >> 16);
}

__device__ __forceinline__ bf16x16 make_frag(const unsigned short* p0,
                                             const unsigned short* p1) {
    bf16x8 lo = *(const bf16x8*)p0;
    bf16x8 hi = *(const bf16x8*)p1;
    bf16x16 r;
#pragma unroll
    for (int i = 0; i < 8; ++i) { r[i] = lo[i]; r[i + 8] = hi[i]; }
    return r;
}

__device__ __forceinline__ f32x8 wmma_bf16(bf16x16 a, bf16x16 b, f32x8 c) {
    return __builtin_amdgcn_wmma_f32_16x16x32_bf16(
        false, a, false, b, (short)0, c, false, false);
}

// scheduling fence: keep fragment loads issued before the WMMA burst
__device__ __forceinline__ void sched_fence() {
    asm volatile("" ::: "memory");
}

// CDNA5 async LDS copy: each lane moves 16B global -> LDS (ASYNCcnt).
__device__ __forceinline__ void async_copy_b128(void* lds_dst, const void* gsrc) {
    unsigned lds_off = (unsigned)(unsigned long long)lds_dst;
    unsigned long long ga = (unsigned long long)gsrc;
    asm volatile("global_load_async_to_lds_b128 %0, %1, off"
                 :: "v"(lds_off), "v"(ga) : "memory");
}

__device__ __forceinline__ void wait_asynccnt0() {
    asm volatile("s_wait_asynccnt 0x0" ::: "memory");
}

// CDNA5 Tensor Data Mover: DMA one 32(k) x 64(rows) bf16 tile, row stride
// DIN elements, global -> LDS with HW padding (16 DW data + 4 DW pad per row
// == 40-halfword padded LDS rows). Issued once per wave; TENSORcnt tracked.
__device__ __forceinline__ void tdm_load_wtile(unsigned lds_addr, const void* gsrc) {
    unsigned long long ga = (unsigned long long)gsrc;
    u32x4 g0;
    g0[0] = 1u;                                                // count=1 (valid user D#)
    g0[1] = lds_addr;                                          // lds_addr
    g0[2] = (unsigned)ga;                                      // global_addr[31:0]
    g0[3] = ((unsigned)(ga >> 32) & 0x01FFFFFFu) | (2u << 30); // addr[56:32] | type=2
    u32x8 g1;
    g1[0] = (1u << 16)        // data_size = 2 bytes
          | (1u << 20)        // pad_enable
          | (3u << 22)        // pad_interval: 16 DWORDs between pads
          | (3u << 25);       // pad_amount: 4 DWORDs
    g1[1] = (1024u & 0xFFFFu) << 16;                 // tensor_dim0 = 1024 (lo16)
    g1[2] = (1024u >> 16) | ((0x100000u & 0xFFFFu) << 16); // dim0 hi | dim1 lo
    g1[3] = ((0x100000u >> 16) & 0xFFFFu) | (32u << 16);   // dim1 hi | tile_dim0=32
    g1[4] = 64u;                                     // tile_dim1 = 64, tile_dim2 = 0
    g1[5] = 1024u;                                   // tensor_dim0_stride[31:0]
    g1[6] = 0u;                                      // stride0 hi | stride1 lo
    g1[7] = 0u;                                      // stride1 hi
    u32x4 gz = {0u, 0u, 0u, 0u};                     // groups 2/3: unused dims
    asm volatile("tensor_load_to_lds %0, %1, %2, %3"
                 :: "s"(g0), "s"(g1), "s"(gz), "s"(gz)
                 : "memory");
}

// ---------- kernel 1: fp32 -> bf16 convert ----------

__global__ void __launch_bounds__(256) cvt_bf16(const float* __restrict__ src,
                                                unsigned short* __restrict__ dst,
                                                int n) {
    int i = (blockIdx.x * 256 + threadIdx.x) * 4;
    if (i < n) {
        float4 f = *(const float4*)(src + i);
        u16x4 o;
        o[0] = f2bfu(f.x); o[1] = f2bfu(f.y);
        o[2] = f2bfu(f.z); o[3] = f2bfu(f.w);
        *(u16x4*)(dst + i) = o;
    }
}

// ---------- kernel 2: fused QKVR projection GEMM (C = X * W^T) ----------
// grid: (DIN/64, (B*S)/256, 4)   block: 256 (8 waves)
// Each wave: 32(M) x 64(N). Weight tile DMA'd by the TDM into ping-pong LDS.

__global__ void __launch_bounds__(256)
gemm_qkvr(const unsigned short* __restrict__ xb,
          const unsigned short* __restrict__ wb,
          unsigned short* __restrict__ qb,
          unsigned short* __restrict__ kb,
          unsigned short* __restrict__ vb,
          float* __restrict__ rf) {
    __shared__ __align__(16) unsigned short sW[2][64 * 40]; // [o][k], 80B rows

    const int tid  = threadIdx.x;
    const int wave = tid >> 5;
    const int lane = tid & 31;
    const int half = lane >> 4;
    const int r16  = lane & 15;

    const int n0    = blockIdx.x * 64;
    const int m0    = blockIdx.y * 256;
    const int which = blockIdx.z;

    const unsigned short* W = wb + (size_t)which * DIN * DIN;
    const unsigned short* xrow0 = xb + (size_t)(m0 + wave * 32 + r16) * DIN;
    const unsigned short* xrow1 = xrow0 + (size_t)16 * DIN;
    const unsigned short* wtile = W + (size_t)n0 * DIN;   // (row n0, k 0)

    const unsigned lds0 = (unsigned)(unsigned long long)&sW[0][0];
    const unsigned lds1 = (unsigned)(unsigned long long)&sW[1][0];

    f32x8 acc0[4] = {}, acc1[4] = {};

    if (wave == 0) tdm_load_wtile(lds0, wtile);   // prologue: tile 0

    const int NIT = DIN / 32;
    for (int it = 0; it < NIT; ++it) {
        if (wave == 0) __builtin_amdgcn_s_wait_tensorcnt((short)0);
        __syncthreads();   // staged tile visible; previous readers done
        if (wave == 0 && it + 1 < NIT)
            tdm_load_wtile((it & 1) ? lds0 : lds1, wtile + (it + 1) * 32);

        const int k0 = it * 32;
        bf16x16 a0 = make_frag(xrow0 + k0 + half * 8, xrow0 + k0 + half * 8 + 16);
        bf16x16 a1 = make_frag(xrow1 + k0 + half * 8, xrow1 + k0 + half * 8 + 16);
        const unsigned short* sWb = sW[it & 1];
        bf16x16 bfr[4];
#pragma unroll
        for (int nt = 0; nt < 4; ++nt) {
            const unsigned short* wp = &sWb[(nt * 16 + r16) * 40 + half * 8];
            bfr[nt] = make_frag(wp, wp + 16);
        }
        sched_fence();   // all 8 ds_loads issued before the WMMA burst
#pragma unroll
        for (int nt = 0; nt < 4; ++nt)
            acc0[nt] = wmma_bf16(a0, bfr[nt], acc0[nt]);
#pragma unroll
        for (int nt = 0; nt < 4; ++nt)
            acc1[nt] = wmma_bf16(a1, bfr[nt], acc1[nt]);
    }

    // C layout: (vgpr j, lane) -> row = j + 8*half, col = r16
    if (which < 3) {
        unsigned short* out = (which == 0) ? qb : (which == 1) ? kb : vb;
#pragma unroll
        for (int nt = 0; nt < 4; ++nt)
#pragma unroll
            for (int j = 0; j < 8; ++j) {
                size_t r0 = (size_t)(m0 + wave * 32 + j + 8 * half);
                int c = n0 + nt * 16 + r16;
                out[r0 * DIN + c]        = f2bfu(acc0[nt][j]);
                out[(r0 + 16) * DIN + c] = f2bfu(acc1[nt][j]);
            }
    } else {
#pragma unroll
        for (int nt = 0; nt < 4; ++nt)
#pragma unroll
            for (int j = 0; j < 8; ++j) {
                size_t r0 = (size_t)(m0 + wave * 32 + j + 8 * half);
                int c = n0 + nt * 16 + r16;
                rf[r0 * DIN + c]        = acc0[nt][j];
                rf[(r0 + 16) * DIN + c] = acc1[nt][j];
            }
    }
}

// ---------- kernel 3: flash attention + fused residual-relu ----------
// grid: (S/128, B*H)   block: 256 (8 waves, 16 q-rows each)

__global__ void __launch_bounds__(256)
attn_kernel(const unsigned short* __restrict__ qb,
            const unsigned short* __restrict__ kb,
            const unsigned short* __restrict__ vb,
            float* __restrict__ rf) {
    __shared__ __align__(16) unsigned short sK[32 * 72];      // [key][d], 144B rows
    __shared__ __align__(16) unsigned short sV[64 * 40];      // [d][key] transposed
    __shared__ __align__(16) unsigned short sP[8 * 16 * 40];  // per-wave P scratch

    const int tid  = threadIdx.x;
    const int wave = tid >> 5;
    const int lane = tid & 31;
    const int half = lane >> 4;
    const int r16  = lane & 15;

    const int bh = blockIdx.y;
    const int b  = bh >> 4;
    const int h  = bh & 15;
    const int q0 = blockIdx.x * 128 + wave * 16;

    const unsigned short* qrow =
        qb + ((size_t)(b * S_ + q0 + r16) * H_ + h) * DH;
    bf16x16 qa0 = make_frag(qrow + half * 8,      qrow + half * 8 + 16);
    bf16x16 qa1 = make_frag(qrow + 32 + half * 8, qrow + 32 + half * 8 + 16);

    f32x8 acc[4] = {};
    float mrow[8], lrow[8];
#pragma unroll
    for (int j = 0; j < 8; ++j) { mrow[j] = -3.0e38f; lrow[j] = 0.f; }

    const float scale = 0.125f; // 1/sqrt(64)

    const int srow = tid >> 3;          // staging: key row 0..31
    const int sdc  = (tid & 7) * 8;     // staging: d chunk

    for (int kv0 = 0; kv0 < S_; kv0 += 32) {
        __syncthreads();   // previous tile's readers done
        // K tile: async global -> LDS, row-major [key][d]
        async_copy_b128(&sK[srow * 72 + sdc],
                        kb + ((size_t)(b * S_ + kv0 + srow) * H_ + h) * DH + sdc);
        // V tile: register transpose -> [d][key]
        {
            const unsigned short* vsrc =
                vb + ((size_t)(b * S_ + kv0 + srow) * H_ + h) * DH + sdc;
            bf16x8 vv = *(const bf16x8*)vsrc;
#pragma unroll
            for (int i = 0; i < 8; ++i)
                sV[(sdc + i) * 40 + srow] = ((const unsigned short*)&vv)[i];
        }
        wait_asynccnt0();
        __syncthreads();

        // scores: two 16x16 tiles, K-dim = 64 -> 2 chained WMMAs each
        const unsigned short* kp0 = &sK[r16 * 72 + half * 8];
        const unsigned short* kp1 = &sK[(16 + r16) * 72 + half * 8];
        bf16x16 kf00 = make_frag(kp0,      kp0 + 16);
        bf16x16 kf01 = make_frag(kp0 + 32, kp0 + 48);
        bf16x16 kf10 = make_frag(kp1,      kp1 + 16);
        bf16x16 kf11 = make_frag(kp1 + 32, kp1 + 48);
        sched_fence();
        f32x8 s0 = {}, s1 = {};
        s0 = wmma_bf16(qa0, kf00, s0);
        s1 = wmma_bf16(qa0, kf10, s1);
        s0 = wmma_bf16(qa1, kf01, s0);
        s1 = wmma_bf16(qa1, kf11, s1);

        // online softmax (row = j + 8*half; reduce over 16 lanes of own half)
        float p0v[8], p1v[8];
#pragma unroll
        for (int j = 0; j < 8; ++j) {
            float a0 = s0[j] * scale, a1 = s1[j] * scale;
            float mx = fmaxf(a0, a1);
#pragma unroll
            for (int m = 1; m < 16; m <<= 1) mx = fmaxf(mx, __shfl_xor(mx, m));
            float mn   = fmaxf(mrow[j], mx);
            float corr = __expf(mrow[j] - mn);
            float p0   = __expf(a0 - mn);
            float p1   = __expf(a1 - mn);
            float rs   = p0 + p1;
#pragma unroll
            for (int m = 1; m < 16; m <<= 1) rs += __shfl_xor(rs, m);
            lrow[j] = lrow[j] * corr + rs;
            mrow[j] = mn;
#pragma unroll
            for (int n = 0; n < 4; ++n) acc[n][j] *= corr;
            p0v[j] = p0; p1v[j] = p1;
        }

        // C-layout P -> A-layout via per-wave LDS (in-order per wave, no barrier)
        unsigned short* pbase = &sP[wave * 640];
#pragma unroll
        for (int j = 0; j < 8; ++j) {
            pbase[(j + 8 * half) * 40 + r16]      = f2bfu(p0v[j]);
            pbase[(j + 8 * half) * 40 + 16 + r16] = f2bfu(p1v[j]);
        }
        const unsigned short* pr = pbase + r16 * 40 + half * 8;
        bf16x16 pa = make_frag(pr, pr + 16);

        // O += P(16x32) x V(32x64): 4 WMMAs along d
        bf16x16 vfr[4];
#pragma unroll
        for (int n = 0; n < 4; ++n) {
            const unsigned short* vp = &sV[(n * 16 + r16) * 40 + half * 8];
            vfr[n] = make_frag(vp, vp + 16);
        }
        sched_fence();
#pragma unroll
        for (int n = 0; n < 4; ++n)
            acc[n] = wmma_bf16(pa, vfr[n], acc[n]);
    }

    // epilogue: normalize, fuse relu(attn + x@Wr^T) in-place over rf
#pragma unroll
    for (int j = 0; j < 8; ++j) {
        float inv = 1.0f / lrow[j];
        size_t r = (size_t)(b * S_ + q0 + j + 8 * half);
#pragma unroll
        for (int n = 0; n < 4; ++n) {
            size_t idx = r * DIN + (h * DH + n * 16 + r16);
            float v = acc[n][j] * inv + rf[idx];
            rf[idx] = fmaxf(v, 0.f);
        }
    }
}

// ---------- kernel 4: LayerNorm over last dim ----------
// grid: B*S  block: 256

__global__ void __launch_bounds__(256)
ln_kernel(const float* __restrict__ rf,
          const float* __restrict__ gamma,
          const float* __restrict__ beta,
          float* __restrict__ out) {
    __shared__ float redS[8], redQ[8];
    const int tid  = threadIdx.x;
    const int wave = tid >> 5;
    const int lane = tid & 31;
    const float* src = rf + (size_t)blockIdx.x * DIN;

    float v[4], s = 0.f, sq = 0.f;
#pragma unroll
    for (int i = 0; i < 4; ++i) {
        v[i] = src[tid + i * 256];
        s += v[i]; sq += v[i] * v[i];
    }
#pragma unroll
    for (int m = 1; m < 32; m <<= 1) {
        s  += __shfl_xor(s, m);
        sq += __shfl_xor(sq, m);
    }
    if (lane == 0) { redS[wave] = s; redQ[wave] = sq; }
    __syncthreads();
    float ts = 0.f, tq = 0.f;
#pragma unroll
    for (int w = 0; w < 8; ++w) { ts += redS[w]; tq += redQ[w]; }
    float mu   = ts * (1.f / DIN);
    float var  = tq * (1.f / DIN) - mu * mu;
    float rstd = rsqrtf(var + 1e-5f);

    float* dst = out + (size_t)blockIdx.x * DIN;
#pragma unroll
    for (int i = 0; i < 4; ++i) {
        int e = tid + i * 256;
        dst[e] = (v[i] - mu) * rstd * gamma[e] + beta[e];
    }
}

// ---------- launch ----------

extern "C" void kernel_launch(void* const* d_in, const int* in_sizes, int n_in,
                              void* d_out, int out_size, void* d_ws, size_t ws_size,
                              hipStream_t stream) {
    const float* x     = (const float*)d_in[0];
    const float* Wq    = (const float*)d_in[1];
    const float* Wk    = (const float*)d_in[2];
    const float* Wv    = (const float*)d_in[3];
    const float* Wr    = (const float*)d_in[4];
    const float* gamma = (const float*)d_in[5];
    const float* beta  = (const float*)d_in[6];
    float* out = (float*)d_out;

    char* ws = (char*)d_ws;
    const size_t MB = 1024ull * 1024ull;
    unsigned short* xb = (unsigned short*)(ws);            // 16 MB
    unsigned short* wb = (unsigned short*)(ws + 16 * MB);  //  8 MB
    unsigned short* qb = (unsigned short*)(ws + 24 * MB);  // 16 MB
    unsigned short* kb = (unsigned short*)(ws + 40 * MB);  // 16 MB
    unsigned short* vb = (unsigned short*)(ws + 56 * MB);  // 16 MB
    float*          rf = (float*)(ws + 72 * MB);           // 32 MB

    const int nx = B_ * S_ * DIN;   // 8,388,608
    const int nw = DIN * DIN;       // 1,048,576

    cvt_bf16<<<nx / 1024, 256, 0, stream>>>(x,  xb, nx);
    cvt_bf16<<<nw / 1024, 256, 0, stream>>>(Wq, wb + 0 * (size_t)nw, nw);
    cvt_bf16<<<nw / 1024, 256, 0, stream>>>(Wk, wb + 1 * (size_t)nw, nw);
    cvt_bf16<<<nw / 1024, 256, 0, stream>>>(Wv, wb + 2 * (size_t)nw, nw);
    cvt_bf16<<<nw / 1024, 256, 0, stream>>>(Wr, wb + 3 * (size_t)nw, nw);

    gemm_qkvr<<<dim3(DIN / 64, (B_ * S_) / 256, 4), 256, 0, stream>>>(
        xb, wb, qb, kb, vb, rf);

    attn_kernel<<<dim3(S_ / 128, B_ * H_), 256, 0, stream>>>(qb, kb, vb, rf);

    ln_kernel<<<B_ * S_, 256, 0, stream>>>(rf, gamma, beta, out);
}
